// GLMAttention_28527172780499
// MI455X (gfx1250) — compile-verified, matrix operand order checked by reading, not run on hardware
//
#include <hip/hip_runtime.h>
#include <stdint.h>
#include <math.h>

// ---------------------------------------------------------------------------
// Types
// ---------------------------------------------------------------------------
typedef __bf16 bf16;
typedef __attribute__((ext_vector_type(16))) __bf16 v16bf;
typedef __attribute__((ext_vector_type(8)))  float  v8f;
typedef __attribute__((ext_vector_type(4)))  unsigned int u32x4;
typedef __attribute__((ext_vector_type(4)))  int i32x4;
typedef __attribute__((ext_vector_type(8)))  int i32x8;

struct frag_bits { u32x4 lo, hi; };

#ifndef __has_builtin
#define __has_builtin(x) 0
#endif

#if __has_builtin(__builtin_amdgcn_tensor_load_to_lds) && \
    __has_builtin(__builtin_amdgcn_s_wait_tensorcnt)
#define USE_TDM 1
#else
#define USE_TDM 0
#endif

__device__ __forceinline__ bf16 f2bf(float f) {
  unsigned int u = __builtin_bit_cast(unsigned int, f);
  unsigned int r = u + 0x7FFFu + ((u >> 16) & 1u);   // round-to-nearest-even
  unsigned short h = (unsigned short)(r >> 16);
  return __builtin_bit_cast(bf16, h);
}

// Load a 16x32 bf16 A/B fragment slice for this lane: two 16-byte loads.
// CDNA5 16-bit matrix layout: lanes 0-15 -> row = lane, K runs {0..7, 16..23};
// lanes 16-31 -> row = lane-16, K runs {8..15, 24..31}.
// Caller passes p = &row_base[half*8]; we read p[0..7] and p[16..23].
__device__ __forceinline__ v16bf load_frag(const bf16* p) {
  frag_bits fb;
  fb.lo = *(const u32x4*)(p);
  fb.hi = *(const u32x4*)(p + 16);
  return __builtin_bit_cast(v16bf, fb);
}

__device__ __forceinline__ v8f wmma_bf16(v16bf a, v16bf b, v8f c) {
  return __builtin_amdgcn_wmma_f32_16x16x32_bf16(
      /*neg_a=*/false, a, /*neg_b=*/false, b,
      /*c_mod=*/(short)0, c, /*reuse_a=*/false, /*reuse_b=*/false);
}

#if USE_TDM
// Low 32 bits of the generic (LDS-aperture) address == raw LDS byte offset.
__device__ __forceinline__ unsigned lds_addr_of(const void* p) {
  return (unsigned)(unsigned long long)p;
}

// TDM: DMA a 128-row x 32-col bf16 tile (row stride = strideElems elements)
// from global memory into LDS at ldsOff.  D# per CDNA5 ISA ch.8:
//   group0: count=1 | lds_addr | global_addr[31:0] | global_addr[56:32]+type=2
//   group1: data_size=1(2B), tensor_dim0=32, tensor_dim1=128,
//           tile_dim0=32, tile_dim1=128, tensor_dim0_stride=strideElems
//   groups 2/3: zero (2-D tensor, tile_dim2=0).
__device__ __forceinline__ void tdm_load_tile_128x32(const bf16* gptr,
                                                     unsigned ldsOff,
                                                     unsigned strideElems) {
  unsigned long long ga = (unsigned long long)(const void*)gptr;
  u32x4 g0;
  g0[0] = 1u;                                            // count=1 (valid)
  g0[1] = ldsOff;                                        // lds_addr
  g0[2] = (unsigned)ga;                                  // global_addr[31:0]
  g0[3] = (unsigned)((ga >> 32) & 0x1FFFFFFull) | (2u << 30);  // hi + type=2
  i32x8 g1;
  g1[0] = (int)(1u << 16);          // workgroup_mask=0, data_size=1 (2 bytes)
  g1[1] = (int)(32u << 16);         // tensor_dim0[15:0]=32 in [31:16]
  g1[2] = (int)(128u << 16);        // tensor_dim0 hi=0, tensor_dim1 lo=128
  g1[3] = (int)(32u << 16);         // tensor_dim1 hi=0, tile_dim0=32
  g1[4] = (int)128u;                // tile_dim1=128, tile_dim2=0
  g1[5] = (int)strideElems;         // tensor_dim0_stride[31:0]
  g1[6] = 0;                        // stride hi, tensor_dim1_stride lo
  g1[7] = 0;
  i32x4 z4 = {0, 0, 0, 0};
#if __clang_major__ >= 23
  i32x8 z8 = {0, 0, 0, 0, 0, 0, 0, 0};
  __builtin_amdgcn_tensor_load_to_lds(g0, g1, z4, z4, z8, 0);
#else
  __builtin_amdgcn_tensor_load_to_lds(g0, g1, z4, z4, 0);
#endif
}
#endif  // USE_TDM

// ---------------------------------------------------------------------------
// 1. fp32 -> bf16 conversion (grid-stride)
// ---------------------------------------------------------------------------
__global__ void f32_to_bf16_kernel(const float* __restrict__ in,
                                   bf16* __restrict__ out, long long n) {
  long long i = (long long)blockIdx.x * blockDim.x + threadIdx.x;
  long long stride = (long long)gridDim.x * blockDim.x;
  for (; i < n; i += stride) out[i] = f2bf(in[i]);
}

// ---------------------------------------------------------------------------
// 2. TN GEMM with bias:  C[m,n] = sum_k A[m,k]*B[n,k] + bias[n]
//    A: MxK bf16 row-major, B: NxK bf16 row-major, C: MxN f32.
//    Block: 256 threads = 8 waves; block tile 128x128; wave tile 32x64.
//    Tiles staged via the Tensor Data Mover (double-buffered LDS), falling
//    back to manual global->LDS staging if the TDM builtin is unavailable.
// ---------------------------------------------------------------------------
__global__ void __launch_bounds__(256)
gemm_tn_bias(const bf16* __restrict__ A, const bf16* __restrict__ B,
             const float* __restrict__ bias, float* __restrict__ C,
             int M, int N, int K) {
  __shared__ bf16 As[2][128 * 32];
  __shared__ bf16 Bs[2][128 * 32];

  const int tid  = threadIdx.x;
  const int lane = tid & 31;
  const int wave = tid >> 5;
  const int waveM = wave & 3;   // 4 waves along M (32 rows each)
  const int waveN = wave >> 2;  // 2 waves along N (64 cols each)
  const int half = lane >> 4;
  const int l16  = lane & 15;

  const int mBase = blockIdx.y * 128;
  const int nBase = blockIdx.x * 128;

  v8f c[2][4];
#pragma unroll
  for (int i = 0; i < 2; i++)
#pragma unroll
    for (int j = 0; j < 4; j++)
#pragma unroll
      for (int r = 0; r < 8; r++) c[i][j][r] = 0.0f;

#if USE_TDM
  // ---- TDM double-buffered pipeline --------------------------------------
  const bf16* Ab = A + (size_t)mBase * K;
  const bf16* Bb = B + (size_t)nBase * K;
  if (wave == 0) {
    tdm_load_tile_128x32(Ab, lds_addr_of(As[0]), (unsigned)K);
    tdm_load_tile_128x32(Bb, lds_addr_of(Bs[0]), (unsigned)K);
  }
  const int nTiles = K / 32;
  for (int t = 0; t < nTiles; ++t) {
    const int buf = t & 1;
    if (wave == 0) {
      if (t + 1 < nTiles) {
        // Prefetch next tile into the other buffer, then wait for current
        // tile only (TDM ops complete in order; 2 newer ops may remain).
        tdm_load_tile_128x32(Ab + (size_t)(t + 1) * 32,
                             lds_addr_of(As[buf ^ 1]), (unsigned)K);
        tdm_load_tile_128x32(Bb + (size_t)(t + 1) * 32,
                             lds_addr_of(Bs[buf ^ 1]), (unsigned)K);
        __builtin_amdgcn_s_wait_tensorcnt(2);
      } else {
        __builtin_amdgcn_s_wait_tensorcnt(0);
      }
    }
    __syncthreads();   // current buffer ready for all waves

    v16bf af[2], bfr[4];
#pragma unroll
    for (int i = 0; i < 2; i++)
      af[i] = load_frag(As[buf] + (waveM * 32 + i * 16 + l16) * 32 + half * 8);
#pragma unroll
    for (int j = 0; j < 4; j++)
      bfr[j] = load_frag(Bs[buf] + (waveN * 64 + j * 16 + l16) * 32 + half * 8);

#pragma unroll
    for (int i = 0; i < 2; i++)
#pragma unroll
      for (int j = 0; j < 4; j++)
        c[i][j] = wmma_bf16(af[i], bfr[j], c[i][j]);

    __syncthreads();   // all reads done before this buffer is DMA'd again
  }
#else
  // ---- Fallback: manual global->LDS staging (single buffer) --------------
  const int lrow = tid >> 1;
  const int lcol = (tid & 1) * 16;
  for (int k0 = 0; k0 < K; k0 += 32) {
    const u32x4* ga = (const u32x4*)(A + (size_t)(mBase + lrow) * K + k0 + lcol);
    const u32x4* gb = (const u32x4*)(B + (size_t)(nBase + lrow) * K + k0 + lcol);
    u32x4 a0 = ga[0];
    u32x4 b0 = gb[0];
    __syncthreads();
    *(u32x4*)(As[0] + lrow * 32 + lcol) = a0;
    *(u32x4*)(Bs[0] + lrow * 32 + lcol) = b0;
    __syncthreads();

    v16bf af[2], bfr[4];
#pragma unroll
    for (int i = 0; i < 2; i++)
      af[i] = load_frag(As[0] + (waveM * 32 + i * 16 + l16) * 32 + half * 8);
#pragma unroll
    for (int j = 0; j < 4; j++)
      bfr[j] = load_frag(Bs[0] + (waveN * 64 + j * 16 + l16) * 32 + half * 8);

#pragma unroll
    for (int i = 0; i < 2; i++)
#pragma unroll
      for (int j = 0; j < 4; j++)
        c[i][j] = wmma_bf16(af[i], bfr[j], c[i][j]);
  }
#endif

  // Store: C layout -> VGPR r holds row (half*8 + r), lane%16 is the column.
#pragma unroll
  for (int j = 0; j < 4; j++) {
    const int col = nBase + waveN * 64 + j * 16 + l16;
    const float bv = bias ? bias[col] : 0.0f;
#pragma unroll
    for (int i = 0; i < 2; i++) {
      const int rowb = mBase + waveM * 32 + i * 16 + half * 8;
#pragma unroll
      for (int r = 0; r < 8; r++)
        C[(size_t)(rowb + r) * N + col] = c[i][j][r] + bv;
    }
  }
}

// ---------------------------------------------------------------------------
// 3. RoPE + scale + relayout.
//    qkv: [S][H*384] f32 (per head: q[128] | k[128] | v[128])
//    Q,K -> bf16 [H][S][128] ; V -> bf16 transposed [H][128][S]
//    Q also pre-scaled by 1/(sqrt(128)*coeff); logits later scaled by coeff.
// ---------------------------------------------------------------------------
__global__ void rope_kernel(const float* __restrict__ qkv,
                            const float* __restrict__ f0r, const float* __restrict__ f0i,
                            const float* __restrict__ f1r, const float* __restrict__ f1i,
                            bf16* __restrict__ Qb, bf16* __restrict__ Kb,
                            bf16* __restrict__ Vt) {
  const int s = blockIdx.x;     // 0..2047
  const int h = blockIdx.y;     // 0..31
  const int e = threadIdx.x;    // 0..31 (== E)
  const float a0 = f0r[s * 32 + e], b0 = f0i[s * 32 + e];
  const float a1 = f1r[s * 32 + e], b1 = f1i[s * 32 + e];
  const float qs = 1.0f / (sqrtf(128.0f) * 4.0f);

  const float* base = qkv + (size_t)s * 12288 + h * 384;
  bf16* Qrow = Qb + ((size_t)h * 2048 + s) * 128;
  bf16* Krow = Kb + ((size_t)h * 2048 + s) * 128;

  // Q rope (scaled)
  float xr = base[e], xi = base[32 + e];
  float yr = base[64 + e], yi = base[96 + e];
  Qrow[e]      = f2bf((xr * a0 - xi * b0) * qs);
  Qrow[32 + e] = f2bf((xr * b0 + xi * a0) * qs);
  Qrow[64 + e] = f2bf((yr * a1 - yi * b1) * qs);
  Qrow[96 + e] = f2bf((yr * b1 + yi * a1) * qs);

  // K rope
  xr = base[128 + e]; xi = base[160 + e];
  yr = base[192 + e]; yi = base[224 + e];
  Krow[e]      = f2bf(xr * a0 - xi * b0);
  Krow[32 + e] = f2bf(xr * b0 + xi * a0);
  Krow[64 + e] = f2bf(yr * a1 - yi * b1);
  Krow[96 + e] = f2bf(yr * b1 + yi * a1);

  // V transpose: Vt[h][d][s]
#pragma unroll
  for (int j = 0; j < 4; j++) {
    const int d = e + j * 32;
    Vt[((size_t)h * 128 + d) * 2048 + s] = f2bf(base[256 + d]);
  }
}

// ---------------------------------------------------------------------------
// 4. Causal flash attention, one 16-row Q tile per wave, 32 keys per step.
//    Scores: 8 WMMAs/32-key block; PV: 8 WMMAs/32-key block.
//    Output: bf16 [S][H*128] (feeds the output-projection GEMM as its A).
// ---------------------------------------------------------------------------
__global__ void __launch_bounds__(256)
attn_kernel(const bf16* __restrict__ Qb, const bf16* __restrict__ Kb,
            const bf16* __restrict__ Vt, bf16* __restrict__ Ob) {
  __shared__ bf16 Pld[8][16 * 32];   // per-wave P bounce tile (1 KB each)

  const int tid  = threadIdx.x;
  const int lane = tid & 31;
  const int wave = tid >> 5;
  const int half = lane >> 4;
  const int l16  = lane & 15;
  const int h  = blockIdx.y;
  const int qb = blockIdx.x * 128 + wave * 16;
  const float coeff = 4.0f;  // LAYER_IDX + 1

  const bf16* Qh = Qb + (size_t)h * 2048 * 128;
  const bf16* Kh = Kb + (size_t)h * 2048 * 128;
  const bf16* Vh = Vt + (size_t)h * 128 * 2048;

  // Q fragments for this wave's 16 rows, d split into 4 chunks of 32.
  v16bf aq[4];
#pragma unroll
  for (int t = 0; t < 4; t++)
    aq[t] = load_frag(Qh + (size_t)(qb + l16) * 128 + t * 32 + half * 8);

  v8f o[8];
  float m[8], l[8], alpha[8];
#pragma unroll
  for (int cidx = 0; cidx < 8; cidx++)
#pragma unroll
    for (int r = 0; r < 8; r++) o[cidx][r] = 0.0f;
#pragma unroll
  for (int r = 0; r < 8; r++) { m[r] = -1e30f; l[r] = 0.0f; }

  const int q_hi = qb + 15;
  for (int kj = 0; kj <= q_hi; kj += 32) {
    v8f s0, s1;
#pragma unroll
    for (int r = 0; r < 8; r++) { s0[r] = 0.0f; s1[r] = 0.0f; }

#pragma unroll
    for (int t = 0; t < 4; t++) {
      v16bf bk0 = load_frag(Kh + (size_t)(kj + l16) * 128 + t * 32 + half * 8);
      s0 = wmma_bf16(aq[t], bk0, s0);
    }
#pragma unroll
    for (int t = 0; t < 4; t++) {
      v16bf bk1 = load_frag(Kh + (size_t)(kj + 16 + l16) * 128 + t * 32 + half * 8);
      s1 = wmma_bf16(aq[t], bk1, s1);
    }

    // Online softmax. Row r of this half lives in s*[r] across the 16-lane
    // group; columns are kj+l16 and kj+16+l16.
#pragma unroll
    for (int r = 0; r < 8; r++) {
      const int row = qb + half * 8 + r;
      const int c0 = kj + l16;
      const int c1 = kj + 16 + l16;
      float v0 = (c0 <= row) ? s0[r] * coeff : -1e30f;
      float v1 = (c1 <= row) ? s1[r] * coeff : -1e30f;
      float rmax = fmaxf(v0, v1);
      rmax = fmaxf(rmax, __shfl_xor(rmax, 1));
      rmax = fmaxf(rmax, __shfl_xor(rmax, 2));
      rmax = fmaxf(rmax, __shfl_xor(rmax, 4));
      rmax = fmaxf(rmax, __shfl_xor(rmax, 8));
      const float mn = fmaxf(m[r], rmax);
      alpha[r] = __expf(m[r] - mn);
      const float p0 = __expf(v0 - mn);
      const float p1 = __expf(v1 - mn);
      float rs = p0 + p1;
      rs += __shfl_xor(rs, 1);
      rs += __shfl_xor(rs, 2);
      rs += __shfl_xor(rs, 4);
      rs += __shfl_xor(rs, 8);
      l[r] = l[r] * alpha[r] + rs;
      m[r] = mn;
      Pld[wave][(half * 8 + r) * 32 + l16]      = f2bf(p0);
      Pld[wave][(half * 8 + r) * 32 + 16 + l16] = f2bf(p1);
    }

    // Rescale running output by alpha (row-indexed, same as accumulators).
#pragma unroll
    for (int cidx = 0; cidx < 8; cidx++)
#pragma unroll
      for (int r = 0; r < 8; r++) o[cidx][r] *= alpha[r];

    // Re-read P as a 16x32 A-fragment (wave-internal LDS bounce; in-order DS).
    v16bf pf = load_frag(&Pld[wave][l16 * 32 + half * 8]);

    // PV: B[n=d][k=key] from transposed V.
#pragma unroll
    for (int cidx = 0; cidx < 8; cidx++) {
      v16bf bv = load_frag(Vh + (size_t)(cidx * 16 + l16) * 2048 + kj + half * 8);
      o[cidx] = wmma_bf16(pf, bv, o[cidx]);
    }
  }

  // Normalize by l and store bf16 at [row][h*128 + d].
#pragma unroll
  for (int cidx = 0; cidx < 8; cidx++) {
#pragma unroll
    for (int r = 0; r < 8; r++) {
      const int row = qb + half * 8 + r;
      const int d = cidx * 16 + l16;
      Ob[(size_t)row * 4096 + h * 128 + d] = f2bf(o[cidx][r] / l[r]);
    }
  }
}

// ---------------------------------------------------------------------------
// Launch
// ---------------------------------------------------------------------------
extern "C" void kernel_launch(void* const* d_in, const int* in_sizes, int n_in,
                              void* d_out, int out_size, void* d_ws, size_t ws_size,
                              hipStream_t stream) {
  const float* x    = (const float*)d_in[0];
  const float* f0r  = (const float*)d_in[1];
  const float* f0i  = (const float*)d_in[2];
  const float* f1r  = (const float*)d_in[3];
  const float* f1i  = (const float*)d_in[4];
  /* mask d_in[5] unused: causal mask computed inline */
  const float* wqkv = (const float*)d_in[6];
  const float* bqkv = (const float*)d_in[7];
  const float* wo   = (const float*)d_in[8];
  const float* bo   = (const float*)d_in[9];
  float* out = (float*)d_out;

  const int S = 2048, D = 4096, H = 32, TD = 3 * 4096;

  char* ws = (char*)d_ws;
  size_t off = 0;
  bf16* xb    = (bf16*)(ws + off); off += (size_t)S * D * 2;        // 16 MB
  bf16* wqkvb = (bf16*)(ws + off); off += (size_t)TD * D * 2;       // 96 MB
  bf16* wob   = (bf16*)(ws + off); off += (size_t)D * D * 2;        // 32 MB
  float* qkvf = (float*)(ws + off); off += (size_t)S * TD * 4;      // 96 MB
  bf16* Qb    = (bf16*)(ws + off); off += (size_t)H * S * 128 * 2;  // 16 MB
  bf16* Kb    = (bf16*)(ws + off); off += (size_t)H * S * 128 * 2;  // 16 MB
  bf16* Vt    = (bf16*)(ws + off); off += (size_t)H * 128 * S * 2;  // 16 MB
  bf16* attnb = (bf16*)(ws + off); off += (size_t)S * D * 2;        // 16 MB

  // 1) downconvert to bf16 (weights then live in 192 MB L2 across GEMM tiles)
  f32_to_bf16_kernel<<<1024, 256, 0, stream>>>(x,    xb,    (long long)S * D);
  f32_to_bf16_kernel<<<4096, 256, 0, stream>>>(wqkv, wqkvb, (long long)TD * D);
  f32_to_bf16_kernel<<<2048, 256, 0, stream>>>(wo,   wob,   (long long)D * D);

  // 2) qkv = x @ w_qkv^T + b_qkv   (M=2048, N=12288, K=4096)
  gemm_tn_bias<<<dim3(TD / 128, S / 128), 256, 0, stream>>>(
      xb, wqkvb, bqkv, qkvf, S, TD, D);

  // 3) RoPE + scale + relayout into Q/K head-major, V transposed
  rope_kernel<<<dim3(S, H), 32, 0, stream>>>(qkvf, f0r, f0i, f1r, f1i, Qb, Kb, Vt);

  // 4) causal flash attention -> bf16 [S][D]
  attn_kernel<<<dim3(S / 128, H), 256, 0, stream>>>(Qb, Kb, Vt, attnb);

  // 5) out = attn @ w_o^T + b_o   (M=2048, N=4096, K=4096)
  gemm_tn_bias<<<dim3(D / 128, S / 128), 256, 0, stream>>>(
      attnb, wob, bo, out, S, D, D);
}
